// Attention_42107859370601
// MI455X (gfx1250) — matrix operand order repaired and checked by reading, not compiled, not run on hardware
//
#include <hip/hip_runtime.h>

typedef __bf16 bf16_t;
typedef __attribute__((ext_vector_type(16))) __bf16 v16bf;
typedef __attribute__((ext_vector_type(8)))  float  v8f;
typedef int v4i __attribute__((vector_size(16)));
typedef __attribute__((address_space(1))) v4i* as1_v4i_ptr;
typedef __attribute__((address_space(3))) v4i* as3_v4i_ptr;

#define SEQ    2048
#define DH     64
#define NHEAD  8
#define NB     4
#define DMODEL 512
#define INNER  512
#define VPAD   80      // padded LDS row stride (bf16) for staged tiles: 160B = 10x16B

#ifndef __has_builtin
#define __has_builtin(x) 0
#endif
#if __has_builtin(__builtin_amdgcn_global_load_async_to_lds_b128)
#define ASYNC_LDS 1
#else
#define ASYNC_LDS 0
#endif

union FragU { v16bf v; uint4 u[2]; };

// A operand (16x32 bf16, row-major source [*, ld]):
// lane L (<16): row L, K chunks {0..7, 16..23}; lane L+16: row L, K {8..15, 24..31}
__device__ __forceinline__ v16bf load_frag_A(const bf16_t* base, int ld, int lane) {
    const int row = lane & 15;
    const int kb  = (lane >> 4) << 3;           // 0 or 8
    const bf16_t* p = base + (size_t)row * ld + kb;
    FragU f;
    f.u[0] = *reinterpret_cast<const uint4*>(p);
    f.u[1] = *reinterpret_cast<const uint4*>(p + 16);
    return f.v;
}

// B operand (32x16 bf16) from a [N][K] (transposed) source:
// lane L (<16): column L, K {0..7, 16..23}; lane L+16: column L, K {8..15, 24..31}
__device__ __forceinline__ v16bf load_frag_B(const bf16_t* base, int ld, int lane) {
    const int col = lane & 15;
    const int kb  = (lane >> 4) << 3;           // 0 or 8
    const bf16_t* p = base + (size_t)col * ld + kb;
    FragU f;
    f.u[0] = *reinterpret_cast<const uint4*>(p);
    f.u[1] = *reinterpret_cast<const uint4*>(p + 16);
    return f.v;
}

__device__ __forceinline__ v8f wmma_bf16(v16bf a, v16bf b, v8f c) {
    return __builtin_amdgcn_wmma_f32_16x16x32_bf16(false, a, false, b, (short)0, c,
                                                   false, false);
}

// Intra-wave LDS ordering (per-wave DS ops complete in order) + compiler fence.
__device__ __forceinline__ void lds_fence() {
    asm volatile("s_wait_dscnt 0x0" ::: "memory");
}

// 16B global -> LDS copy primitive for cooperative staging.
__device__ __forceinline__ void stage16(const bf16_t* g, bf16_t* l) {
#if ASYNC_LDS
    __builtin_amdgcn_global_load_async_to_lds_b128(
        (as1_v4i_ptr)(g), (as3_v4i_ptr)(l), 0, 0);
#else
    *reinterpret_cast<uint4*>(l) = *reinterpret_cast<const uint4*>(g);
#endif
}

// ---------------- prep: fp32 -> bf16 convert / transpose-convert ----------------
__global__ void cvt_f32_bf16(const float* __restrict__ in, bf16_t* __restrict__ out, int n) {
    int i = blockIdx.x * blockDim.x + threadIdx.x;
    if (i < n) out[i] = (bf16_t)in[i];
}

// out[n][k] = in[k][n]  (in: [K][N] row-major fp32 -> out: [N][K] bf16)
__global__ void transpose_cvt(const float* __restrict__ in, bf16_t* __restrict__ out,
                              int K, int N) {
    int i = blockIdx.x * blockDim.x + threadIdx.x;
    if (i < K * N) {
        int k = i / N, n = i % N;
        out[(size_t)n * K + k] = (bf16_t)in[i];
    }
}

// ---------------- QKV projection GEMM: [8192,512] x [512,1536] ----------------
__global__ __launch_bounds__(256)
void qkv_gemm(const bf16_t* __restrict__ A, const bf16_t* __restrict__ BT,
              bf16_t* __restrict__ Qo, bf16_t* __restrict__ Ko, bf16_t* __restrict__ Vt) {
    const int lane  = threadIdx.x & 31;
    const int wave  = threadIdx.x >> 5;
    const int mbase = blockIdx.y * 128 + wave * 16;
    const int nbase = blockIdx.x * 64;

    const bf16_t* Ab  = A + (size_t)mbase * DMODEL;
    const bf16_t* Bb0 = BT + (size_t)(nbase +  0) * DMODEL;
    const bf16_t* Bb1 = BT + (size_t)(nbase + 16) * DMODEL;
    const bf16_t* Bb2 = BT + (size_t)(nbase + 32) * DMODEL;
    const bf16_t* Bb3 = BT + (size_t)(nbase + 48) * DMODEL;

    v8f c0 = {}, c1 = {}, c2 = {}, c3 = {};
    // software pipeline: fragments for k, prefetch of k+32 into distinct registers
    v16bf a  = load_frag_A(Ab,  DMODEL, lane);
    v16bf b0 = load_frag_B(Bb0, DMODEL, lane);
    v16bf b1 = load_frag_B(Bb1, DMODEL, lane);
    v16bf b2 = load_frag_B(Bb2, DMODEL, lane);
    v16bf b3 = load_frag_B(Bb3, DMODEL, lane);

    for (int k = 0; k < DMODEL; k += 32) {
        const int kn = (k + 32) & (DMODEL - 1);           // wrap: last prefetch harmless
        __builtin_prefetch(Ab  + (size_t)(lane & 15) * DMODEL + ((k + 64) & (DMODEL - 1)), 0, 3);
        __builtin_prefetch(Bb0 + (size_t)(lane & 15) * DMODEL + ((k + 64) & (DMODEL - 1)), 0, 3);
        v16bf an  = load_frag_A(Ab  + kn, DMODEL, lane);
        v16bf bn0 = load_frag_B(Bb0 + kn, DMODEL, lane);
        v16bf bn1 = load_frag_B(Bb1 + kn, DMODEL, lane);
        v16bf bn2 = load_frag_B(Bb2 + kn, DMODEL, lane);
        v16bf bn3 = load_frag_B(Bb3 + kn, DMODEL, lane);

        c0 = wmma_bf16(a, b0, c0);
        c1 = wmma_bf16(a, b1, c1);
        c2 = wmma_bf16(a, b2, c2);
        c3 = wmma_bf16(a, b3, c3);

        a = an; b0 = bn0; b1 = bn1; b2 = bn2; b3 = bn3;
    }

    const int which = blockIdx.x >> 3;   // 0=q 1=k 2=v
    const int hi    = blockIdx.x & 7;    // head
#pragma unroll
    for (int r = 0; r < 8; ++r) {
        int m  = mbase + r + ((lane >> 4) << 3);
        int bi = m >> 11;
        int ti = m & (SEQ - 1);
        float vals[4] = {c0[r], c1[r], c2[r], c3[r]};
        size_t bhn = (size_t)bi * NHEAD + hi;
#pragma unroll
        for (int t = 0; t < 4; ++t) {
            int di = t * 16 + (lane & 15);
            bf16_t v = (bf16_t)vals[t];
            if (which == 0)      Qo[(bhn * SEQ + ti) * DH + di] = v;
            else if (which == 1) Ko[(bhn * SEQ + ti) * DH + di] = v;
            else                 Vt[(bhn * DH + di) * SEQ + ti] = v;  // V transposed
        }
    }
}

// ---------------- rotary (GPT-NeoX halves) in place on Q and K ----------------
__global__ __launch_bounds__(256)
void rotary_inplace(bf16_t* __restrict__ Q, bf16_t* __restrict__ K,
                    const float* __restrict__ pos) {
    int gid = blockIdx.x * 256 + threadIdx.x;
    int row = gid >> 5;                 // b*h*n rows
    int d   = gid & 31;
    int ti  = row & (SEQ - 1);
    const float* pr = pos + (size_t)ti * DH;
    float c1 = __cosf(pr[d]),      s1 = __sinf(pr[d]);
    float c2 = __cosf(pr[d + 32]), s2 = __sinf(pr[d + 32]);

    bf16_t* q = Q + (size_t)row * DH;
    float x1 = (float)q[d], x2 = (float)q[d + 32];
    q[d]      = (bf16_t)(x1 * c1 - x2 * s1);
    q[d + 32] = (bf16_t)(x2 * c2 + x1 * s2);

    bf16_t* kk = K + (size_t)row * DH;
    x1 = (float)kk[d]; x2 = (float)kk[d + 32];
    kk[d]      = (bf16_t)(x1 * c1 - x2 * s1);
    kk[d + 32] = (bf16_t)(x2 * c2 + x1 * s2);
}

// ---------------- flash attention: one (b,h) per blockIdx.y, 128 q-rows/block ----------------
// K/V tiles staged cooperatively into double-buffered LDS via async global->LDS
// copies (ASYNCcnt) so the 8 waves share one copy instead of 8x redundant VMEM.
__global__ __launch_bounds__(256)
void attn_flash(const bf16_t* __restrict__ Q, const bf16_t* __restrict__ K,
                const bf16_t* __restrict__ Vt, const unsigned char* __restrict__ mask,
                bf16_t* __restrict__ ctx) {
    __shared__ __align__(16) bf16_t ldsK[2][32][VPAD];   // [key][d],   10 KB
    __shared__ __align__(16) bf16_t ldsV[2][64][VPAD];   // [d][key],   20 KB
    __shared__ __align__(16) bf16_t ldsP[8][16][48];     // per-wave P relayout, 12 KB

    const int tid   = threadIdx.x;
    const int lane  = tid & 31;
    const int wave  = tid >> 5;
    const int bh    = blockIdx.y;
    const int bi    = bh >> 3;
    const int hi    = bh & 7;
    const int qbase = blockIdx.x * 128 + wave * 16;

    const bf16_t* Qb = Q  + (size_t)bh * SEQ * DH;
    const bf16_t* Kb = K  + (size_t)bh * SEQ * DH;
    const bf16_t* Vb = Vt + (size_t)bh * DH * SEQ;
    const unsigned char* mb = mask + (size_t)bi * SEQ;

    // staging assignments (one b128 per thread per tile)
    const int krow = tid >> 3, kch = tid & 7;   // K tile: 32 rows x 8 chunks
    const int vrow = tid >> 2, vch = tid & 3;   // V tile: 64 rows x 4 chunks

    // Q fragments (16 rows x 64 d) held in registers for the whole key loop
    v16bf qa0 = load_frag_A(Qb + (size_t)qbase * DH,      DH, lane);  // d 0..31
    v16bf qa1 = load_frag_A(Qb + (size_t)qbase * DH + 32, DH, lane);  // d 32..63

    v8f o0 = {}, o1 = {}, o2 = {}, o3 = {};
    float mrow[8], lrow[8];
#pragma unroll
    for (int r = 0; r < 8; ++r) { mrow[r] = -3.3895314e38f; lrow[r] = 0.f; }

    const float scale  = 0.125f;           // 64^-0.5
    const float NEGMAX = -3.4028235e38f;

    // preload buffer 0 with the j=0 tiles
    stage16(Kb + (size_t)krow * DH + kch * 8,       &ldsK[0][krow][kch * 8]);
    stage16(Vb + (size_t)vrow * SEQ + vch * 8,      &ldsV[0][vrow][vch * 8]);

    for (int j0 = 0, it = 0; j0 < SEQ; j0 += 32, ++it) {
        const int cur = it & 1;
        const int nxt = cur ^ 1;
        const int jn  = (j0 + 32) & (SEQ - 1);     // wrap: last prefetch harmless

        __syncthreads();   // everyone done reading buf[nxt] from the previous round
        __builtin_prefetch(Kb + (size_t)(jn + 32) * DH, 0, 3);
        __builtin_prefetch(Vb + (size_t)jn + 64, 0, 3);
        stage16(Kb + (size_t)(jn + krow) * DH + kch * 8, &ldsK[nxt][krow][kch * 8]);
        stage16(Vb + (size_t)vrow * SEQ + jn + vch * 8,  &ldsV[nxt][vrow][vch * 8]);
#if ASYNC_LDS
        // own previous batch (filling buf[cur]) complete; newest 2 still in flight
        asm volatile("s_wait_asynccnt 0x2" ::: "memory");
#else
        lds_fence();
#endif
        __syncthreads();   // all waves' buf[cur] data visible

        const int col = lane & 15;
        const bool mk0 = mb[j0 + col]      != 0;
        const bool mk1 = mb[j0 + 16 + col] != 0;

        // S tiles from LDS-staged K: keys j0..+15 / j0+16..+31, K-dim = 64
        v16bf kf0 = load_frag_B(&ldsK[cur][0][0],   VPAD, lane);
        v16bf kf1 = load_frag_B(&ldsK[cur][0][32],  VPAD, lane);
        v16bf kf2 = load_frag_B(&ldsK[cur][16][0],  VPAD, lane);
        v16bf kf3 = load_frag_B(&ldsK[cur][16][32], VPAD, lane);
        v8f s0 = {}, s1 = {};
        s0 = wmma_bf16(qa0, kf0, s0);
        s0 = wmma_bf16(qa1, kf1, s0);
        s1 = wmma_bf16(qa0, kf2, s1);
        s1 = wmma_bf16(qa1, kf3, s1);

        // V fragments (consumed only after softmax -> latency hidden)
        v16bf vf0 = load_frag_B(&ldsV[cur][0][0],  VPAD, lane);
        v16bf vf1 = load_frag_B(&ldsV[cur][16][0], VPAD, lane);
        v16bf vf2 = load_frag_B(&ldsV[cur][32][0], VPAD, lane);
        v16bf vf3 = load_frag_B(&ldsV[cur][48][0], VPAD, lane);

        float p0[8], p1[8];
#pragma unroll
        for (int r = 0; r < 8; ++r) {
            float a = mk0 ? s0[r] * scale : NEGMAX;
            float b = mk1 ? s1[r] * scale : NEGMAX;
            // row max: row elements live across one 16-lane half -> xor butterfly 1,2,4,8
            float tm = fmaxf(a, b);
            tm = fmaxf(tm, __shfl_xor(tm, 1, 32));
            tm = fmaxf(tm, __shfl_xor(tm, 2, 32));
            tm = fmaxf(tm, __shfl_xor(tm, 4, 32));
            tm = fmaxf(tm, __shfl_xor(tm, 8, 32));
            float mnew = fmaxf(mrow[r], tm);
            float corr = __expf(mrow[r] - mnew);
            float e0 = mk0 ? __expf(a - mnew) : 0.f;
            float e1 = mk1 ? __expf(b - mnew) : 0.f;
            float rs = e0 + e1;
            rs += __shfl_xor(rs, 1, 32);
            rs += __shfl_xor(rs, 2, 32);
            rs += __shfl_xor(rs, 4, 32);
            rs += __shfl_xor(rs, 8, 32);
            lrow[r] = lrow[r] * corr + rs;
            mrow[r] = mnew;
            o0[r] *= corr; o1[r] *= corr; o2[r] *= corr; o3[r] *= corr;
            p0[r] = e0; p1[r] = e1;
        }

        // Relayout P (C layout) -> A layout through per-wave-private LDS tile.
        const int prowbase = (lane >> 4) << 3;
#pragma unroll
        for (int r = 0; r < 8; ++r) {
            ldsP[wave][prowbase + r][col]      = (bf16_t)p0[r];
            ldsP[wave][prowbase + r][col + 16] = (bf16_t)p1[r];
        }
        lds_fence();
        v16bf pa = load_frag_A(&ldsP[wave][0][0], 48, lane);

        // O += P(16x32) · V(32x64)
        o0 = wmma_bf16(pa, vf0, o0);
        o1 = wmma_bf16(pa, vf1, o1);
        o2 = wmma_bf16(pa, vf2, o2);
        o3 = wmma_bf16(pa, vf3, o3);
    }

    // normalize and write context [b][n][h*d] (bf16, A-ready for output GEMM)
#pragma unroll
    for (int r = 0; r < 8; ++r) {
        float inv = lrow[r] > 0.f ? 1.f / lrow[r] : 0.f;
        int q = qbase + ((lane >> 4) << 3) + r;
        bf16_t* dst = ctx + ((size_t)(bi * SEQ + q)) * INNER + hi * DH + (lane & 15);
        dst[0]  = (bf16_t)(o0[r] * inv);
        dst[16] = (bf16_t)(o1[r] * inv);
        dst[32] = (bf16_t)(o2[r] * inv);
        dst[48] = (bf16_t)(o3[r] * inv);
    }
}

// ---------------- output projection: [8192,512] x [512,512] + bias, fp32 out ----------------
__global__ __launch_bounds__(256)
void out_gemm(const bf16_t* __restrict__ A, const bf16_t* __restrict__ BT,
              const float* __restrict__ bias, float* __restrict__ out) {
    const int lane  = threadIdx.x & 31;
    const int wave  = threadIdx.x >> 5;
    const int mbase = blockIdx.y * 128 + wave * 16;
    const int nbase = blockIdx.x * 64;

    const bf16_t* Ab  = A + (size_t)mbase * INNER;
    const bf16_t* Bb0 = BT + (size_t)(nbase +  0) * INNER;
    const bf16_t* Bb1 = BT + (size_t)(nbase + 16) * INNER;
    const bf16_t* Bb2 = BT + (size_t)(nbase + 32) * INNER;
    const bf16_t* Bb3 = BT + (size_t)(nbase + 48) * INNER;

    v8f c0 = {}, c1 = {}, c2 = {}, c3 = {};
    v16bf a  = load_frag_A(Ab,  INNER, lane);
    v16bf b0 = load_frag_B(Bb0, INNER, lane);
    v16bf b1 = load_frag_B(Bb1, INNER, lane);
    v16bf b2 = load_frag_B(Bb2, INNER, lane);
    v16bf b3 = load_frag_B(Bb3, INNER, lane);

    for (int k = 0; k < INNER; k += 32) {
        const int kn = (k + 32) & (INNER - 1);
        __builtin_prefetch(Ab + (size_t)(lane & 15) * INNER + ((k + 64) & (INNER - 1)), 0, 3);
        v16bf an  = load_frag_A(Ab  + kn, INNER, lane);
        v16bf bn0 = load_frag_B(Bb0 + kn, INNER, lane);
        v16bf bn1 = load_frag_B(Bb1 + kn, INNER, lane);
        v16bf bn2 = load_frag_B(Bb2 + kn, INNER, lane);
        v16bf bn3 = load_frag_B(Bb3 + kn, INNER, lane);

        c0 = wmma_bf16(a, b0, c0);
        c1 = wmma_bf16(a, b1, c1);
        c2 = wmma_bf16(a, b2, c2);
        c3 = wmma_bf16(a, b3, c3);

        a = an; b0 = bn0; b1 = bn1; b2 = bn2; b3 = bn3;
    }
#pragma unroll
    for (int r = 0; r < 8; ++r) {
        int m = mbase + r + ((lane >> 4) << 3);
        float vals[4] = {c0[r], c1[r], c2[r], c3[r]};
#pragma unroll
        for (int t = 0; t < 4; ++t) {
            int colg = nbase + t * 16 + (lane & 15);
            out[(size_t)m * DMODEL + colg] = vals[t] + bias[colg];
        }
    }
}

extern "C" void kernel_launch(void* const* d_in, const int* in_sizes, int n_in,
                              void* d_out, int out_size, void* d_ws, size_t ws_size,
                              hipStream_t stream) {
    const float*         x    = (const float*)d_in[0];
    const unsigned char* mask = (const unsigned char*)d_in[1];
    const float*         pos  = (const float*)d_in[2];
    const float*         Wqkv = (const float*)d_in[3];
    const float*         Wout = (const float*)d_in[4];
    const float*         bout = (const float*)d_in[5];
    float* out = (float*)d_out;

    char* ws = (char*)d_ws;
    size_t off = 0;
    bf16_t* xb    = (bf16_t*)(ws + off); off += (size_t)NB * SEQ * DMODEL * 2;     // 8 MB
    bf16_t* wqkvT = (bf16_t*)(ws + off); off += (size_t)3 * INNER * DMODEL * 2;    // 1.5 MB
    bf16_t* woutT = (bf16_t*)(ws + off); off += (size_t)DMODEL * INNER * 2;        // 0.5 MB
    bf16_t* Qb    = (bf16_t*)(ws + off); off += (size_t)NB * NHEAD * SEQ * DH * 2; // 8 MB
    bf16_t* Kb    = (bf16_t*)(ws + off); off += (size_t)NB * NHEAD * SEQ * DH * 2; // 8 MB
    bf16_t* Vt    = (bf16_t*)(ws + off); off += (size_t)NB * NHEAD * DH * SEQ * 2; // 8 MB
    bf16_t* ctx   = (bf16_t*)(ws + off); off += (size_t)NB * SEQ * INNER * 2;      // 8 MB

    cvt_f32_bf16<<<(NB * SEQ * DMODEL + 255) / 256, 256, 0, stream>>>(x, xb, NB * SEQ * DMODEL);
    transpose_cvt<<<(DMODEL * 3 * INNER + 255) / 256, 256, 0, stream>>>(Wqkv, wqkvT, DMODEL, 3 * INNER);
    transpose_cvt<<<(INNER * DMODEL + 255) / 256, 256, 0, stream>>>(Wout, woutT, INNER, DMODEL);

    qkv_gemm<<<dim3(24, 64), 256, 0, stream>>>(xb, wqkvT, Qb, Kb, Vt);
    rotary_inplace<<<(NB * NHEAD * SEQ * 32) / 256, 256, 0, stream>>>(Qb, Kb, pos);
    attn_flash<<<dim3(SEQ / 128, NB * NHEAD), 256, 0, stream>>>(Qb, Kb, Vt, mask, ctx);
    out_gemm<<<dim3(8, 64), 256, 0, stream>>>(ctx, woutT, bout, out);
}
